// ChebGCN_53240414601484
// MI455X (gfx1250) — matrix operand order repaired and checked by reading, not compiled, not run on hardware
//
#include <hip/hip_runtime.h>
#include <math.h>

typedef __attribute__((ext_vector_type(16))) __bf16 v16bf;
typedef __attribute__((ext_vector_type(8)))  float  v8f;

#define IN_F   256
#define HID_F  128
#define E1_F   64
#define E2_F   32
#define CHEB_K 3

__device__ __forceinline__ unsigned short f2bf(float f) {
  unsigned int u = __float_as_uint(f);
  u += 0x7FFFu + ((u >> 16) & 1u);          // round-to-nearest-even
  return (unsigned short)(u >> 16);
}

__global__ void k_zero4(float4* __restrict__ p, long long n4) {
  long long t = (long long)blockIdx.x * blockDim.x + threadIdx.x;
  if (t < n4) p[t] = make_float4(0.f, 0.f, 0.f, 0.f);
}

__global__ void k_deg(const int* __restrict__ src, float* __restrict__ deg, int E) {
  int t = blockIdx.x * blockDim.x + threadIdx.x;
  if (t < E) atomicAdd(&deg[src[t]], 1.0f);
}

__global__ void k_dinv(float* __restrict__ deg, int Nn) {
  int t = blockIdx.x * blockDim.x + threadIdx.x;
  if (t < Nn) { float d = deg[t]; deg[t] = (d > 0.0f) ? rsqrtf(d) : 0.0f; }
}

__global__ void k_norm(const int* __restrict__ src, const int* __restrict__ dst,
                       const float* __restrict__ dinv, float* __restrict__ norm, int E) {
  int t = blockIdx.x * blockDim.x + threadIdx.x;
  if (t < E) norm[t] = -dinv[src[t]] * dinv[dst[t]];
}

// W[rows][cols] fp32 -> Wt[cols][rows] bf16 (B-operand lanes then read contiguously)
__global__ void k_cvtw(const float* __restrict__ W, unsigned short* __restrict__ Wt,
                       int rows, int cols) {
  int t = blockIdx.x * blockDim.x + threadIdx.x;
  if (t >= rows * cols) return;
  int r = t / cols, c = t % cols;
  Wt[(long long)c * rows + r] = f2bf(W[t]);
}

// fp32 -> bf16, vectorized (counts are multiples of 4)
__global__ void k_cvta(const float4* __restrict__ A, uint2* __restrict__ B, long long n4) {
  long long t = (long long)blockIdx.x * blockDim.x + threadIdx.x;
  if (t >= n4) return;
  float4 v = A[t];
  union { unsigned short us[4]; uint2 u; } r;
  r.us[0] = f2bf(v.x); r.us[1] = f2bf(v.y); r.us[2] = f2bf(v.z); r.us[3] = f2bf(v.w);
  B[t] = r.u;
}

// Fused Cheb recurrence + convert: B = bf16(2*P2 - T0)   (T2 never needed in fp32)
__global__ void k_cvta_cheb(const float4* __restrict__ P2, const float4* __restrict__ T0,
                            uint2* __restrict__ B, long long n4) {
  long long t = (long long)blockIdx.x * blockDim.x + threadIdx.x;
  if (t >= n4) return;
  float4 p = P2[t], q = T0[t];
  union { unsigned short us[4]; uint2 u; } r;
  r.us[0] = f2bf(2.0f * p.x - q.x); r.us[1] = f2bf(2.0f * p.y - q.y);
  r.us[2] = f2bf(2.0f * p.z - q.z); r.us[3] = f2bf(2.0f * p.w - q.w);
  B[t] = r.u;
}

// One wave = one 16x16 tile of Out = act(A[M x K] @ W[K x NC] (+bias)).
// A bf16 row-major, W bf16 pre-transposed [NC][K]. K,NC compile-time -> full unroll.
template <int K, int NC>
__global__ void k_gemm_wmma(const unsigned short* __restrict__ Abf,
                            const unsigned short* __restrict__ Wt,
                            const float* __restrict__ bias,
                            float* __restrict__ Out,
                            int accum, int bias_relu) {
  const int lane = threadIdx.x;                       // 0..31
  const int tr   = blockIdx.x;                        // 16-row tile
  const int tc   = blockIdx.y * blockDim.y + threadIdx.y;
  const int hi   = lane >> 4;                         // half-wave select
  const int row  = tr * 16 + (lane & 15);
  const int col  = tc * 16 + (lane & 15);
  const int mb   = hi ? 8 : 0;                        // C/D VGPR M-base (ISA layout)

  const unsigned short* Ar = Abf + (long long)row * K;
  const unsigned short* Bc = Wt + (long long)col * K;

  v8f c;
  if (accum) {
#pragma unroll
    for (int r = 0; r < 8; ++r)
      c[r] = Out[(long long)(tr * 16 + mb + r) * NC + col];
  } else {
#pragma unroll
    for (int r = 0; r < 8; ++r) c[r] = 0.0f;
  }

#pragma unroll
  for (int k0 = 0; k0 < K; k0 += 32) {
    // A 16x32 bf16: lanes 0-15 -> K 0..7 / 16..23 ; lanes 16-31 -> K 8..15 / 24..31
    const int ka = k0 + (hi ? 8 : 0);
    union { v16bf v; uint4 q[2]; } av;
    av.q[0] = *(const uint4*)(Ar + ka);
    av.q[1] = *(const uint4*)(Ar + ka + 16);

    // B 32x16 bf16: lanes 0-15 -> K 0..7 / 16..23 ; lanes 16-31 -> K 8..15 / 24..31 (from Wt[col][k])
    const int kb = k0 + (hi ? 8 : 0);
    union { v16bf v; uint4 q[2]; } bv;
    bv.q[0] = *(const uint4*)(Bc + kb);
    bv.q[1] = *(const uint4*)(Bc + kb + 16);

    c = __builtin_amdgcn_wmma_f32_16x16x32_bf16(false, av.v, false, bv.v,
                                                (short)0, c, false, false);
  }

  float bval = bias_relu ? bias[col] : 0.0f;
#pragma unroll
  for (int r = 0; r < 8; ++r) {
    float v = c[r];
    if (bias_relu) { v += bval; v = v > 0.0f ? v : 0.0f; }
    Out[(long long)(tr * 16 + mb + r) * NC + col] = v;
  }
}

// P[dst] += H[src] * norm. One wave per edge; lane covers F/32 consecutive features.
// Edge index / norm loads are wave-uniform -> scalar path.
template <int F>
__global__ void k_prop(const float* __restrict__ H, const int* __restrict__ src,
                       const int* __restrict__ dst, const float* __restrict__ norm,
                       float* __restrict__ P, int E) {
  int e0 = blockIdx.x * blockDim.y + threadIdx.y;     // uniform within wave
  if (e0 >= E) return;
  const int e = __builtin_amdgcn_readfirstlane(e0);
  const int s = src[e];
  const int d = dst[e];
  const float w = norm[e];
  constexpr int C = F / 32;                           // 4 or 2 floats per lane
  const int fo = (int)threadIdx.x * C;
  const float* hp = H + (long long)s * F + fo;
  float*       pp = P + (long long)d * F + fo;
  if constexpr (C == 4) {
    float4 hv = *(const float4*)hp;
    atomicAdd(pp + 0, hv.x * w);
    atomicAdd(pp + 1, hv.y * w);
    atomicAdd(pp + 2, hv.z * w);
    atomicAdd(pp + 3, hv.w * w);
  } else {
    float2 hv = *(const float2*)hp;
    atomicAdd(pp + 0, hv.x * w);
    atomicAdd(pp + 1, hv.y * w);
  }
}

// logits = h @ W(32x2) + b ; softmax
__global__ void k_head(const float* __restrict__ H, const float* __restrict__ W,
                       const float* __restrict__ b, float* __restrict__ out, int Nn) {
  int t = blockIdx.x * blockDim.x + threadIdx.x;
  if (t >= Nn) return;
  const float* h = H + (long long)t * E2_F;
  float l0 = b[0], l1 = b[1];
#pragma unroll
  for (int k = 0; k < E2_F; ++k) {
    float hv = h[k];
    l0 = fmaf(hv, W[k * 2 + 0], l0);
    l1 = fmaf(hv, W[k * 2 + 1], l1);
  }
  float m = fmaxf(l0, l1);
  float e0 = __expf(l0 - m), e1 = __expf(l1 - m);
  float inv = 1.0f / (e0 + e1);
  out[t * 2 + 0] = e0 * inv;
  out[t * 2 + 1] = e1 * inv;
}

extern "C" void kernel_launch(void* const* d_in, const int* in_sizes, int n_in,
                              void* d_out, int out_size, void* d_ws, size_t ws_size,
                              hipStream_t stream) {
  const float* x       = (const float*)d_in[0];
  const int*   eidx    = (const int*)d_in[1];
  const float* W_lin1  = (const float*)d_in[2];
  const float* b_lin1  = (const float*)d_in[3];
  const float* W_cheb1 = (const float*)d_in[4];
  const float* b_cheb1 = (const float*)d_in[5];
  const float* W_cheb2 = (const float*)d_in[6];
  const float* b_cheb2 = (const float*)d_in[7];
  const float* W_lin2  = (const float*)d_in[8];
  const float* b_lin2  = (const float*)d_in[9];
  float* out = (float*)d_out;

  const int Nn = in_sizes[0] / IN_F;
  const int E  = in_sizes[1] / 2;
  const int* src = eidx;
  const int* dst = eidx + E;

  char* ws = (char*)d_ws;
  size_t off = 0;
  auto alloc = [&](size_t bytes) -> char* {
    char* p = ws + off;
    off += (bytes + 255) & ~(size_t)255;
    return p;
  };
  float*          deg  = (float*)alloc((size_t)Nn * 4);
  float*          norm = (float*)alloc((size_t)E * 4);
  unsigned short* Wt1  = (unsigned short*)alloc((size_t)IN_F * HID_F * 2);
  unsigned short* WtC1 = (unsigned short*)alloc((size_t)CHEB_K * HID_F * E1_F * 2);
  unsigned short* WtC2 = (unsigned short*)alloc((size_t)CHEB_K * E1_F * E2_F * 2);
  float*          h1   = (float*)alloc((size_t)Nn * HID_F * 4);
  float*          P1   = (float*)alloc((size_t)Nn * HID_F * 4);
  float*          P2   = (float*)alloc((size_t)Nn * HID_F * 4);
  float*          o1   = (float*)alloc((size_t)Nn * E1_F * 4);
  float*          o2   = (float*)alloc((size_t)Nn * E2_F * 4);
  unsigned short* Abf  = (unsigned short*)alloc((size_t)Nn * IN_F * 2);  // reused A operand
  (void)ws_size; (void)n_in; (void)out_size;

  const int TB = 256;
  const int Mt = Nn / 16;                 // N = 100000 -> 6250 exact tiles
  dim3 blk4(32, 4), blk2(32, 2), blkE(32, 8);
  const int EB = (E + 7) / 8;             // prop: 8 edges (waves) per block

  // ---- edge normalization ----
  k_zero4<<<(Nn / 4 + TB - 1) / TB, TB, 0, stream>>>((float4*)deg, Nn / 4);
  k_deg  <<<(E + TB - 1) / TB, TB, 0, stream>>>(src, deg, E);
  k_dinv <<<(Nn + TB - 1) / TB, TB, 0, stream>>>(deg, Nn);
  k_norm <<<(E + TB - 1) / TB, TB, 0, stream>>>(src, dst, deg, norm, E);

  // ---- weights -> bf16, transposed ----
  k_cvtw<<<(IN_F * HID_F + TB - 1) / TB, TB, 0, stream>>>(W_lin1, Wt1, IN_F, HID_F);
  for (int k = 0; k < CHEB_K; ++k) {
    k_cvtw<<<(HID_F * E1_F + TB - 1) / TB, TB, 0, stream>>>(
        W_cheb1 + (size_t)k * HID_F * E1_F, WtC1 + (size_t)k * HID_F * E1_F, HID_F, E1_F);
    k_cvtw<<<(E1_F * E2_F + TB - 1) / TB, TB, 0, stream>>>(
        W_cheb2 + (size_t)k * E1_F * E2_F, WtC2 + (size_t)k * E1_F * E2_F, E1_F, E2_F);
  }

  long long nx4 = (long long)Nn * IN_F / 4;
  long long nh4 = (long long)Nn * HID_F / 4;
  long long ne4 = (long long)Nn * E1_F / 4;

  // ---- lin1: h1 = relu(x @ W_lin1 + b) ----
  k_cvta<<<(int)((nx4 + TB - 1) / TB), TB, 0, stream>>>((const float4*)x, (uint2*)Abf, nx4);
  k_gemm_wmma<IN_F, HID_F><<<dim3(Mt, 2), blk4, 0, stream>>>(Abf, Wt1, b_lin1, h1, 0, 1);

  // ---- ChebConv1 (128 -> 64), K=3 ----
  k_cvta<<<(int)((nh4 + TB - 1) / TB), TB, 0, stream>>>((const float4*)h1, (uint2*)Abf, nh4);
  k_gemm_wmma<HID_F, E1_F><<<dim3(Mt, 1), blk4, 0, stream>>>(Abf, WtC1, b_cheb1, o1, 0, 0);
  k_zero4<<<(int)((nh4 + TB - 1) / TB), TB, 0, stream>>>((float4*)P1, nh4);
  k_prop<HID_F><<<EB, blkE, 0, stream>>>(h1, src, dst, norm, P1, E);
  k_cvta<<<(int)((nh4 + TB - 1) / TB), TB, 0, stream>>>((const float4*)P1, (uint2*)Abf, nh4);
  k_gemm_wmma<HID_F, E1_F><<<dim3(Mt, 1), blk4, 0, stream>>>(
      Abf, WtC1 + (size_t)1 * HID_F * E1_F, b_cheb1, o1, 1, 0);
  k_zero4<<<(int)((nh4 + TB - 1) / TB), TB, 0, stream>>>((float4*)P2, nh4);
  k_prop<HID_F><<<EB, blkE, 0, stream>>>(P1, src, dst, norm, P2, E);
  k_cvta_cheb<<<(int)((nh4 + TB - 1) / TB), TB, 0, stream>>>(
      (const float4*)P2, (const float4*)h1, (uint2*)Abf, nh4);
  k_gemm_wmma<HID_F, E1_F><<<dim3(Mt, 1), blk4, 0, stream>>>(
      Abf, WtC1 + (size_t)2 * HID_F * E1_F, b_cheb1, o1, 1, 1);

  // ---- ChebConv2 (64 -> 32), K=3 ----
  k_cvta<<<(int)((ne4 + TB - 1) / TB), TB, 0, stream>>>((const float4*)o1, (uint2*)Abf, ne4);
  k_gemm_wmma<E1_F, E2_F><<<dim3(Mt, 1), blk2, 0, stream>>>(Abf, WtC2, b_cheb2, o2, 0, 0);
  k_zero4<<<(int)((ne4 + TB - 1) / TB), TB, 0, stream>>>((float4*)P1, ne4);
  k_prop<E1_F><<<EB, blkE, 0, stream>>>(o1, src, dst, norm, P1, E);
  k_cvta<<<(int)((ne4 + TB - 1) / TB), TB, 0, stream>>>((const float4*)P1, (uint2*)Abf, ne4);
  k_gemm_wmma<E1_F, E2_F><<<dim3(Mt, 1), blk2, 0, stream>>>(
      Abf, WtC2 + (size_t)1 * E1_F * E2_F, b_cheb2, o2, 1, 0);
  k_zero4<<<(int)((ne4 + TB - 1) / TB), TB, 0, stream>>>((float4*)P2, ne4);
  k_prop<E1_F><<<EB, blkE, 0, stream>>>(P1, src, dst, norm, P2, E);
  k_cvta_cheb<<<(int)((ne4 + TB - 1) / TB), TB, 0, stream>>>(
      (const float4*)P2, (const float4*)o1, (uint2*)Abf, ne4);
  k_gemm_wmma<E1_F, E2_F><<<dim3(Mt, 1), blk2, 0, stream>>>(
      Abf, WtC2 + (size_t)2 * E1_F * E2_F, b_cheb2, o2, 1, 1);

  // ---- head: softmax(o2 @ W_lin2 + b) ----
  k_head<<<(Nn + TB - 1) / TB, TB, 0, stream>>>(o2, W_lin2, b_lin2, out, Nn);
}
